// BoWExtractor_49941879717946
// MI455X (gfx1250) — matrix operand order repaired and checked by reading, not compiled, not run on hardware
//
#include <hip/hip_runtime.h>
#include <hip/hip_fp16.h>
#include <math.h>

typedef __attribute__((ext_vector_type(16))) _Float16 v16h;
typedef __attribute__((ext_vector_type(8)))  _Float16 v8h;
typedef __attribute__((ext_vector_type(4)))  _Float16 v4h;
typedef __attribute__((ext_vector_type(8)))  float    v8f;

#define C_DIM   768
#define N_EMB   4096
#define N_IMG   128
#define L_TOK   196             // 197 tokens, token 0 dropped
#define M_ROWS  (N_IMG * L_TOK) // 25088, multiple of 16
#define SCALE   30.0f           // INV_DELTA / DIST_NORM_PREV
#define EPS     1e-5f

// ---------------------------------------------------------------------------
// Kernel 1: per-row scale s[r] = 30 / max(||x_row||, eps). One block per row.
// ---------------------------------------------------------------------------
__global__ __launch_bounds__(256)
void rowscale_kernel(const float* __restrict__ x, float* __restrict__ s) {
    const int r   = blockIdx.x;
    const int img = r / L_TOK;
    const int tok = 1 + (r % L_TOK);
    const float* xr = x + ((size_t)img * 197 + tok) * C_DIM;

    const int t = threadIdx.x;
    float acc = 0.f;
    for (int k = t; k < C_DIM; k += 256) { float v = xr[k]; acc += v * v; }
    #pragma unroll
    for (int off = 16; off; off >>= 1) acc += __shfl_xor(acc, off, 32);

    __shared__ float sm[8];
    if ((t & 31) == 0) sm[t >> 5] = acc;
    __syncthreads();
    if (t == 0) {
        float tot = 0.f;
        #pragma unroll
        for (int i = 0; i < 8; ++i) tot += sm[i];
        s[r] = SCALE / fmaxf(sqrtf(tot), EPS);
    }
}

// ---------------------------------------------------------------------------
// Kernel 2: embedding f32 -> f16 (rows are unit-norm, values in [-1,1]).
// ---------------------------------------------------------------------------
__global__ __launch_bounds__(256)
void cvt_emb_kernel(const float* __restrict__ e, _Float16* __restrict__ eh) {
    const int i = (blockIdx.x * 256 + threadIdx.x) * 4;   // N_EMB*C_DIM % 1024 == 0
    float4 v = *(const float4*)(e + i);
    v4h h = { (_Float16)v.x, (_Float16)v.y, (_Float16)v.z, (_Float16)v.w };
    *(v4h*)(eh + i) = h;
}

// ---------------------------------------------------------------------------
// Kernel 3: NT GEMM via v_wmma_f32_16x16x32_f16, software-pipelined.
//   logits[r, n] = sum_k x_f16[r,k] * emb_f16[n,k]      (scale applied later)
// Block = 8 waves = 16(M) x 512(N) tile; each wave: 16x64 (4 accumulators).
// K loop double-buffers all operand fragments in registers so the next
// k-step's 12 b128 loads issue before the current 4 WMMAs (wait overlap).
// ---------------------------------------------------------------------------
__device__ __forceinline__ v16h cvt_afrag(v8f a0, v8f a1) {
    v16h af;
    #pragma unroll
    for (int i = 0; i < 8; ++i) {
        af[i]     = (_Float16)a0[i];
        af[8 + i] = (_Float16)a1[i];
    }
    return af;
}

__global__ __launch_bounds__(256)
void gemm_kernel(const float* __restrict__ x, const _Float16* __restrict__ eh,
                 float* __restrict__ logits) {
    const int lane   = threadIdx.x & 31;
    const int wave   = threadIdx.x >> 5;
    const int half   = lane >> 4;         // 0: lanes 0-15, 1: lanes 16-31
    const int lrow   = lane & 15;
    const int m_base = blockIdx.x * 16;
    const int n_base = blockIdx.y * 512 + wave * 64;

    // A source row pointer (f32, converted on the fly; x token 0 skipped)
    const int r   = m_base + lrow;
    const int img = r / L_TOK;
    const int tok = 1 + (r % L_TOK);
    // A layout: lane holds 16 contiguous K at h*16 (matches B)
    const float* arow = x + ((size_t)img * 197 + tok) * C_DIM + half * 16;
    // B layout: lane holds column (lane&15), 16 contiguous K at h*16
    const _Float16* brow = eh + (size_t)(n_base + lrow) * C_DIM + half * 16;

    v8f acc0 = {}, acc1 = {}, acc2 = {}, acc3 = {};

    // ---- prologue: load k-step 0 ----
    v8f  ra0 = *(const v8f*)(arow);
    v8f  ra1 = *(const v8f*)(arow + 8);
    v16h b0  = *(const v16h*)(brow);
    v16h b1  = *(const v16h*)(brow + 16 * C_DIM);
    v16h b2  = *(const v16h*)(brow + 32 * C_DIM);
    v16h b3  = *(const v16h*)(brow + 48 * C_DIM);

    #pragma unroll
    for (int kk = 32; kk < C_DIM; kk += 32) {
        // ---- issue next k-step loads (independent of current WMMAs) ----
        v8f  na0 = *(const v8f*)(arow + kk);
        v8f  na1 = *(const v8f*)(arow + kk + 8);
        const _Float16* bp = brow + kk;
        v16h nb0 = *(const v16h*)(bp);
        v16h nb1 = *(const v16h*)(bp + 16 * C_DIM);
        v16h nb2 = *(const v16h*)(bp + 32 * C_DIM);
        v16h nb3 = *(const v16h*)(bp + 48 * C_DIM);

        // ---- compute on current fragments (loaded one iteration ago) ----
        v16h af = cvt_afrag(ra0, ra1);
        acc0 = __builtin_amdgcn_wmma_f32_16x16x32_f16(false, af, false, b0, (short)0, acc0, false, false);
        acc1 = __builtin_amdgcn_wmma_f32_16x16x32_f16(false, af, false, b1, (short)0, acc1, false, false);
        acc2 = __builtin_amdgcn_wmma_f32_16x16x32_f16(false, af, false, b2, (short)0, acc2, false, false);
        acc3 = __builtin_amdgcn_wmma_f32_16x16x32_f16(false, af, false, b3, (short)0, acc3, false, false);

        // ---- rotate buffers (renamed away by full unroll) ----
        ra0 = na0; ra1 = na1;
        b0 = nb0; b1 = nb1; b2 = nb2; b3 = nb3;
    }

    // ---- epilogue: last k-step ----
    {
        v16h af = cvt_afrag(ra0, ra1);
        acc0 = __builtin_amdgcn_wmma_f32_16x16x32_f16(false, af, false, b0, (short)0, acc0, false, false);
        acc1 = __builtin_amdgcn_wmma_f32_16x16x32_f16(false, af, false, b1, (short)0, acc1, false, false);
        acc2 = __builtin_amdgcn_wmma_f32_16x16x32_f16(false, af, false, b2, (short)0, acc2, false, false);
        acc3 = __builtin_amdgcn_wmma_f32_16x16x32_f16(false, af, false, b3, (short)0, acc3, false, false);
    }

    // C/D layout: VGPR j -> row m_base + j + half*8, col n_base + subtile*16 + lrow
    #pragma unroll
    for (int j = 0; j < 8; ++j) {
        const size_t row = (size_t)(m_base + j + half * 8);
        float* out = logits + row * N_EMB + n_base + lrow;
        out[0]  = acc0[j];
        out[16] = acc1[j];
        out[32] = acc2[j];
        out[48] = acc3[j];
    }
}

// ---------------------------------------------------------------------------
// Kernel 4: fused scale + softmax over 4096, in place. One block per row.
//   codes = softmax(s[r] * logits_row)
// ---------------------------------------------------------------------------
__global__ __launch_bounds__(256)
void softmax_kernel(float* __restrict__ logits, const float* __restrict__ s) {
    const int r = blockIdx.x;
    const float sc = s[r];
    float* row = logits + (size_t)r * N_EMB;
    const int t = threadIdx.x;

    float v[16];
    float mx = -INFINITY;
    #pragma unroll
    for (int i = 0; i < 16; ++i) {
        v[i] = sc * row[t + i * 256];
        mx = fmaxf(mx, v[i]);
    }
    #pragma unroll
    for (int off = 16; off; off >>= 1) mx = fmaxf(mx, __shfl_xor(mx, off, 32));

    __shared__ float sm[8];
    __shared__ float ss[8];
    if ((t & 31) == 0) sm[t >> 5] = mx;
    __syncthreads();
    float bm = sm[0];
    #pragma unroll
    for (int i = 1; i < 8; ++i) bm = fmaxf(bm, sm[i]);

    float sum = 0.f;
    #pragma unroll
    for (int i = 0; i < 16; ++i) { v[i] = __expf(v[i] - bm); sum += v[i]; }
    #pragma unroll
    for (int off = 16; off; off >>= 1) sum += __shfl_xor(sum, off, 32);
    if ((t & 31) == 0) ss[t >> 5] = sum;
    __syncthreads();
    float tot = 0.f;
    #pragma unroll
    for (int i = 0; i < 8; ++i) tot += ss[i];
    const float inv = 1.0f / tot;
    #pragma unroll
    for (int i = 0; i < 16; ++i) row[t + i * 256] = v[i] * inv;
}

// ---------------------------------------------------------------------------
// Kernel 5: bow[n,k] = mean over the inner 10x10 token positions of codes.
// ---------------------------------------------------------------------------
__global__ __launch_bounds__(256)
void bow_mean_kernel(const float* __restrict__ codes, float* __restrict__ bow) {
    const int n = blockIdx.x;
    const int k = blockIdx.y * 256 + threadIdx.x;
    const float* base = codes + (size_t)n * L_TOK * N_EMB + k;
    float acc = 0.f;
    #pragma unroll 5
    for (int i = 0; i < 100; ++i) {
        const int rr = 2 + i / 10;
        const int cc = 2 + i % 10;
        acc += base[(size_t)(rr * 14 + cc) * N_EMB];
    }
    bow[(size_t)n * N_EMB + k] = acc * 0.01f;
}

// ---------------------------------------------------------------------------
// Kernel 6: L1-normalize bow rows in place. One block per image.
// ---------------------------------------------------------------------------
__global__ __launch_bounds__(256)
void bow_norm_kernel(float* __restrict__ bow) {
    const int n = blockIdx.x;
    float* row = bow + (size_t)n * N_EMB;
    const int t = threadIdx.x;

    float v[16];
    float acc = 0.f;
    #pragma unroll
    for (int i = 0; i < 16; ++i) { v[i] = row[t + i * 256]; acc += fabsf(v[i]); }
    #pragma unroll
    for (int off = 16; off; off >>= 1) acc += __shfl_xor(acc, off, 32);

    __shared__ float sm[8];
    if ((t & 31) == 0) sm[t >> 5] = acc;
    __syncthreads();
    float tot = 0.f;
    #pragma unroll
    for (int i = 0; i < 8; ++i) tot += sm[i];
    const float inv = 1.0f / fmaxf(tot, EPS);
    #pragma unroll
    for (int i = 0; i < 16; ++i) row[t + i * 256] = v[i] * inv;
}

// ---------------------------------------------------------------------------
extern "C" void kernel_launch(void* const* d_in, const int* in_sizes, int n_in,
                              void* d_out, int out_size, void* d_ws, size_t ws_size,
                              hipStream_t stream) {
    const float* x   = (const float*)d_in[0];   // (128, 197, 768) f32
    const float* emb = (const float*)d_in[1];   // (4096, 768) f32

    float* bow   = (float*)d_out;                              // 128*4096
    float* codes = (float*)d_out + (size_t)N_IMG * N_EMB;      // 128*196*4096

    // Workspace: f16 embedding (6.29 MB, 32B-aligned) + row scales (100 KB)
    _Float16* eh = (_Float16*)d_ws;
    float*    s  = (float*)((char*)d_ws + (size_t)N_EMB * C_DIM * sizeof(_Float16));

    cvt_emb_kernel <<< (N_EMB * C_DIM) / 1024, 256, 0, stream >>>(emb, eh);
    rowscale_kernel<<< M_ROWS, 256, 0, stream >>>(x, s);

    dim3 ggrid(M_ROWS / 16, N_EMB / 512);       // (1568, 8)
    gemm_kernel    <<< ggrid, 256, 0, stream >>>(x, eh, codes);

    softmax_kernel <<< M_ROWS, 256, 0, stream >>>(codes, s);
    bow_mean_kernel<<< dim3(N_IMG, N_EMB / 256), 256, 0, stream >>>(codes, bow);
    bow_norm_kernel<<< N_IMG, 256, 0, stream >>>(bow);
}
